// Model_71648644432344
// MI455X (gfx1250) — compile-verified, matrix operand order checked by reading, not run on hardware
//
#include <hip/hip_runtime.h>
#include <math.h>

typedef __attribute__((ext_vector_type(16))) _Float16 v16h;
typedef __attribute__((ext_vector_type(8)))  _Float16 v8h;
typedef __attribute__((ext_vector_type(8)))  float    v8f;

#define BB    32
#define LL    512
#define CC    16
#define DM    128
#define DS    16
#define DCONV 4
#define DIN   256      // EXP*DM
#define DTR   8
#define NP    65
#define PRED  96
#define HNF   (DM*NP)  // 8320
#define NSEQ  (BB*CC)  // 512
#define MROWS (NSEQ*NP)// 33280
#define DBCLD 48       // padded ld for (DTR + 2*DS)=40

// ---------------------------------------------------------------------------
// f32 -> f16 bulk convert (weights, once per launch)
// ---------------------------------------------------------------------------
__global__ void k_cvt(const float* __restrict__ src, _Float16* __restrict__ dst, int n) {
  const int i = blockIdx.x * blockDim.x + threadIdx.x;
  if (i < n) dst[i] = (_Float16)src[i];
}

// ---------------------------------------------------------------------------
// RevIN stats + normalize + patch-embedding conv. One block per sequence.
// Emits the patch embedding directly in f16 (only ever consumed as GEMM A).
// ---------------------------------------------------------------------------
__global__ void k_patch(const float* __restrict__ x_enc,
                        const float* __restrict__ pe_w,
                        const float* __restrict__ pe_b,
                        _Float16* __restrict__ p16,
                        float* __restrict__ meanstd) {
  __shared__ float xn[LL];
  __shared__ float red[256];
  const int bc  = blockIdx.x;           // 0..511
  const int b   = bc / CC, c = bc % CC;
  const int tid = threadIdx.x;          // 0..255

  const float* xb = x_enc + ((size_t)b * LL) * CC + c;
  float v0 = xb[(size_t)tid * CC];
  float v1 = xb[(size_t)(tid + 256) * CC];

  red[tid] = v0 + v1;
  __syncthreads();
  for (int s = 128; s > 0; s >>= 1) { if (tid < s) red[tid] += red[tid + s]; __syncthreads(); }
  const float mean = red[0] * (1.0f / LL);
  __syncthreads();
  const float d0 = v0 - mean, d1 = v1 - mean;
  red[tid] = d0 * d0 + d1 * d1;
  __syncthreads();
  for (int s = 128; s > 0; s >>= 1) { if (tid < s) red[tid] += red[tid + s]; __syncthreads(); }
  const float stdv = sqrtf(red[0] * (1.0f / LL) + 1e-5f);
  const float inv  = 1.0f / stdv;
  xn[tid]       = d0 * inv;
  xn[tid + 256] = d1 * inv;
  if (tid == 0) { meanstd[bc] = mean; meanstd[NSEQ + bc] = stdv; }
  __syncthreads();

  _Float16* prow = p16 + (size_t)bc * NP * DM;
  for (int idx = tid; idx < NP * DM; idx += 256) {
    const int np = idx / DM, dm = idx % DM;
    float acc = pe_b[dm];
    const int t0 = np * 8 - 8;           // stride 8, left pad 8
#pragma unroll
    for (int k = 0; k < 16; k++) {
      const int t = t0 + k;
      if (t >= 0 && t < LL) acc += pe_w[dm * 16 + k] * xn[t];
    }
    prow[idx] = (_Float16)acc;
  }
}

// ---------------------------------------------------------------------------
// f16 operand fetch matching the 16-bit 16x32 WMMA VGPR layout:
// 8 contiguous halfs at +0 and 8 at +16 (K-halves picked per lane).
// Two 16B loads, no converts.
// ---------------------------------------------------------------------------
__device__ __forceinline__ v16h ldfrag(const _Float16* __restrict__ p) {
  const v8h lo = *(const v8h*)(p);
  const v8h hi = *(const v8h*)(p + 16);
  v16h r;
#pragma unroll
  for (int i = 0; i < 8; i++) { r[i] = lo[i]; r[i + 8] = hi[i]; }
  return r;
}

// ---------------------------------------------------------------------------
// O[M,N] = A[M,K] * Bw[N,K]^T via v_wmma_f32_16x16x32_f16.
// One wave32 per 16x64 output strip (4 accumulators): A fragment fetched once
// per K-step, 4 back-to-back independent WMMAs. All operands pre-converted
// f16 -> inner loop is 10 clause-able b128 loads + 4 WMMAs, zero VALU cvt.
// Ragged N handled by clamping B rows into range (masked at the store),
// so control flow is uniform and EXEC is all-1s at every WMMA.
// ---------------------------------------------------------------------------
template <typename OutT>
__global__ void k_gemm(const _Float16* __restrict__ A, const _Float16* __restrict__ Bw,
                       OutT* __restrict__ O, int N, int K,
                       int lda, int ldb, int ldo, int stripsN, int totalStrips) {
  const int strip = blockIdx.x * (blockDim.x >> 5) + (threadIdx.x >> 5);
  if (strip >= totalStrips) return;                  // wave-uniform exit
  const int tm = strip / stripsN, sn = strip % stripsN;
  const int lane  = threadIdx.x & 31;
  const int r     = lane & 15;
  const int khalf = (lane >> 4) << 3;                // 0 or 8

  const _Float16* arow = A + (size_t)(tm * 16 + r) * lda + khalf;
  const _Float16* brow[4];
#pragma unroll
  for (int j = 0; j < 4; j++) {
    int bi = sn * 64 + j * 16 + r;
    if (bi > N - 1) bi = N - 1;                      // clamp: masked at store
    brow[j] = Bw + (size_t)bi * ldb + khalf;
  }

  v8f acc0 = {}, acc1 = {}, acc2 = {}, acc3 = {};
  for (int k = 0; k < K; k += 32) {
    const v16h a  = ldfrag(arow + k);
    const v16h b0 = ldfrag(brow[0] + k);
    const v16h b1 = ldfrag(brow[1] + k);
    const v16h b2 = ldfrag(brow[2] + k);
    const v16h b3 = ldfrag(brow[3] + k);
    acc0 = __builtin_amdgcn_wmma_f32_16x16x32_f16(false, a, false, b0, (short)0, acc0, false, false);
    acc1 = __builtin_amdgcn_wmma_f32_16x16x32_f16(false, a, false, b1, (short)0, acc1, false, false);
    acc2 = __builtin_amdgcn_wmma_f32_16x16x32_f16(false, a, false, b2, (short)0, acc2, false, false);
    acc3 = __builtin_amdgcn_wmma_f32_16x16x32_f16(false, a, false, b3, (short)0, acc3, false, false);
  }

  const int n   = lane & 15;               // D: column = lane&15
  const int mhi = (lane >> 4) * 8;         // rows v + (lane>>4)*8
  OutT* obase = O + (size_t)(tm * 16 + mhi) * ldo + sn * 64 + n;
  v8f accs[4] = {acc0, acc1, acc2, acc3};
#pragma unroll
  for (int j = 0; j < 4; j++) {
    if (sn * 64 + j * 16 + n < N) {
#pragma unroll
      for (int v = 0; v < 8; v++) obase[(size_t)v * ldo + j * 16] = (OutT)accs[j][v];
    }
  }
}

// ---------------------------------------------------------------------------
// causal depthwise conv over patch dim + SiLU. One thread per (m,d).
// Writes f32 (for the scan) and f16 (GEMM A operand of x_proj).
// ---------------------------------------------------------------------------
__global__ void k_conv(const float* __restrict__ xz, const float* __restrict__ cw,
                       const float* __restrict__ cb, float* __restrict__ xc,
                       _Float16* __restrict__ xc16, int total) {
  const int idx = blockIdx.x * blockDim.x + threadIdx.x;
  if (idx >= total) return;
  const int d = idx % DIN, m = idx / DIN;
  const int bc = m / NP, pp = m % NP;
  float acc = cb[d];
#pragma unroll
  for (int j = 0; j < DCONV; j++) {
    const int t = pp - (DCONV - 1) + j;
    if (t >= 0) acc += cw[d * DCONV + j] * xz[((size_t)(bc * NP + t)) * (2 * DIN) + d];
  }
  const float s = acc * (1.0f / (1.0f + expf(-acc)));
  xc[idx]   = s;
  xc16[idx] = (_Float16)s;
}

// ---------------------------------------------------------------------------
// dt = softplus(dt_raw @ dtp_w^T + dtp_b). One thread per (m,d), K=8.
// ---------------------------------------------------------------------------
__global__ void k_dt(const float* __restrict__ dbc, const float* __restrict__ dtw,
                     const float* __restrict__ dtb, float* __restrict__ dtv, int total) {
  const int idx = blockIdx.x * blockDim.x + threadIdx.x;
  if (idx >= total) return;
  const int d = idx % DIN, m = idx / DIN;
  float acc = dtb[d];
#pragma unroll
  for (int rr = 0; rr < DTR; rr++) acc += dbc[(size_t)m * DBCLD + rr] * dtw[d * DTR + rr];
  dtv[idx] = (acc > 20.0f) ? acc : log1pf(expf(acc));
}

// ---------------------------------------------------------------------------
// selective scan: one thread per (sequence, channel), 16-wide SSM state in
// registers, sequential over NP=65 patches. Fuses +x*D and *silu(z).
// Emits y directly in f16 (only ever consumed as GEMM A of out_proj).
// ---------------------------------------------------------------------------
__global__ void k_scan(const float* __restrict__ dtv, const float* __restrict__ dbc,
                       const float* __restrict__ xc,  const float* __restrict__ xz,
                       const float* __restrict__ A_log, const float* __restrict__ Dp,
                       _Float16* __restrict__ y16) {
  const int idx = blockIdx.x * blockDim.x + threadIdx.x;
  if (idx >= NSEQ * DIN) return;
  const int d = idx % DIN, bc = idx / DIN;

  float Av[DS];
#pragma unroll
  for (int s = 0; s < DS; s++) Av[s] = -expf(A_log[d * DS + s]);
  float h[DS];
#pragma unroll
  for (int s = 0; s < DS; s++) h[s] = 0.0f;
  const float Dd = Dp[d];

  for (int pp = 0; pp < NP; pp++) {
    const size_t m  = (size_t)bc * NP + pp;
    const float dt  = dtv[m * DIN + d];
    const float xv  = xc[m * DIN + d];
    const float* Bm = dbc + m * DBCLD + DTR;
    const float* Cm = Bm + DS;
    const float dtx = dt * xv;
    float acc = 0.0f;
#pragma unroll
    for (int s = 0; s < DS; s++) {
      h[s] = expf(dt * Av[s]) * h[s] + dtx * Bm[s];
      acc += h[s] * Cm[s];
    }
    float yv = acc + xv * Dd;
    const float zv = xz[m * (2 * DIN) + DIN + d];
    yv *= zv * (1.0f / (1.0f + expf(-zv)));   // y * silu(z)
    y16[m * DIN + d] = (_Float16)yv;
  }
}

// ---------------------------------------------------------------------------
// out[b, pred, c] = head[bc, pred] * std[bc] + mean[bc]
// ---------------------------------------------------------------------------
__global__ void k_final(const float* __restrict__ head, const float* __restrict__ meanstd,
                        float* __restrict__ out) {
  const int idx = blockIdx.x * blockDim.x + threadIdx.x;
  if (idx >= BB * PRED * CC) return;
  const int c  = idx % CC;
  const int pr = (idx / CC) % PRED;
  const int b  = idx / (CC * PRED);
  const int bc = b * CC + c;
  out[idx] = head[(size_t)bc * PRED + pr] * meanstd[NSEQ + bc] + meanstd[bc];
}

// ---------------------------------------------------------------------------
template <typename OutT>
static inline void launch_gemm(const _Float16* A, const _Float16* Bw, OutT* O,
                               int M, int N, int K, int lda, int ldb, int ldo,
                               hipStream_t stream) {
  const int tilesM  = (M + 15) / 16;
  const int stripsN = (N + 63) / 64;            // 4 N-tiles per wave
  const int total   = tilesM * stripsN;
  const int wpb     = 8;                        // 8 wave32 per block
  const int blocks  = (total + wpb - 1) / wpb;
  k_gemm<OutT><<<blocks, wpb * 32, 0, stream>>>(A, Bw, O, N, K, lda, ldb, ldo, stripsN, total);
}

extern "C" void kernel_launch(void* const* d_in, const int* in_sizes, int n_in,
                              void* d_out, int out_size, void* d_ws, size_t ws_size,
                              hipStream_t stream) {
  (void)in_sizes; (void)n_in; (void)out_size; (void)ws_size;
  const float* x_enc = (const float*)d_in[0];
  // d_in[1] x_mark_enc: unused by the reference computation
  const float* pe_w  = (const float*)d_in[2];
  const float* pe_b  = (const float*)d_in[3];
  const float* in_w  = (const float*)d_in[4];   // (2, 512, 128)
  const float* cw    = (const float*)d_in[5];   // (2, 256, 4)
  const float* cbi   = (const float*)d_in[6];   // (2, 256)
  const float* xp_w  = (const float*)d_in[7];   // (2, 40, 256)
  const float* dtw   = (const float*)d_in[8];   // (2, 256, 8)
  const float* dtb   = (const float*)d_in[9];   // (2, 256)
  const float* A_log = (const float*)d_in[10];  // (2, 256, 16)
  const float* Dp    = (const float*)d_in[11];  // (2, 256)
  const float* ow    = (const float*)d_in[12];  // (2, 128, 256)
  const float* hw    = (const float*)d_in[13];  // (96, 8320)
  float* out = (float*)d_out;

  // ---- workspace carve-up (f32 region first, then f16 region) ----
  float* ws = (float*)d_ws;
  size_t off = 0;
  float* xz    = ws + off; off += (size_t)MROWS * 2 * DIN;   // in_proj output
  float* xc    = ws + off; off += (size_t)MROWS * DIN;       // conv+silu (f32, for scan)
  float* dbc   = ws + off; off += (size_t)MROWS * DBCLD;     // x_proj output (ld 48)
  float* dtv   = ws + off; off += (size_t)MROWS * DIN;       // softplus(dt)
  float* headb = ws + off; off += (size_t)NSEQ * PRED;       // head output
  float* ms    = ws + off; off += (size_t)2 * NSEQ;          // mean / std

  _Float16* h16 = (_Float16*)(ws + off);
  size_t hoff = 0;
  _Float16* p16   = h16 + hoff; hoff += (size_t)MROWS * DM;   // embeddings / out_proj dst
  _Float16* xc16  = h16 + hoff; hoff += (size_t)MROWS * DIN;  // conv+silu (GEMM A)
  _Float16* y16   = h16 + hoff; hoff += (size_t)MROWS * DIN;  // scan output (GEMM A)
  const int N_INW = 2 * 2 * DIN * DM;           // 131072
  const int N_XPW = 2 * (DTR + 2 * DS) * DIN;   // 20480
  const int N_OW  = 2 * DM * DIN;               // 65536
  const int N_HW  = PRED * HNF;                 // 798720
  _Float16* inw16 = h16 + hoff; hoff += (size_t)N_INW;
  _Float16* xpw16 = h16 + hoff; hoff += (size_t)N_XPW;
  _Float16* ow16  = h16 + hoff; hoff += (size_t)N_OW;
  _Float16* hw16  = h16 + hoff; hoff += (size_t)N_HW;

  // ---- weight conversion (tiny, once per launch) ----
  k_cvt<<<(N_INW + 255) / 256, 256, 0, stream>>>(in_w, inw16, N_INW);
  k_cvt<<<(N_XPW + 255) / 256, 256, 0, stream>>>(xp_w, xpw16, N_XPW);
  k_cvt<<<(N_OW  + 255) / 256, 256, 0, stream>>>(ow,  ow16,  N_OW);
  k_cvt<<<(N_HW  + 255) / 256, 256, 0, stream>>>(hw,  hw16,  N_HW);

  k_patch<<<NSEQ, 256, 0, stream>>>(x_enc, pe_w, pe_b, p16, ms);

  const int totMD = MROWS * DIN;
  const int totSD = NSEQ * DIN;
  for (int i = 0; i < 2; i++) {
    // xz = p @ in_w^T          (M=33280, N=512, K=128)
    launch_gemm<float>(p16, inw16 + (size_t)i * 2 * DIN * DM, xz,
                       MROWS, 2 * DIN, DM, DM, DM, 2 * DIN, stream);
    // xc = silu(depthwise_conv(xz[:, :256]))
    k_conv<<<(totMD + 255) / 256, 256, 0, stream>>>(
        xz, cw + (size_t)i * DIN * DCONV, cbi + (size_t)i * DIN, xc, xc16, totMD);
    // dbc = xc @ xp_w^T        (N=40 -> clamped/guarded strip, ldo=48)
    launch_gemm<float>(xc16, xpw16 + (size_t)i * (DTR + 2 * DS) * DIN, dbc,
                       MROWS, DTR + 2 * DS, DIN, DIN, DIN, DBCLD, stream);
    // dt = softplus(dbc[:, :8] @ dtw^T + dtb)
    k_dt<<<(totMD + 255) / 256, 256, 0, stream>>>(
        dbc, dtw + (size_t)i * DIN * DTR, dtb + (size_t)i * DIN, dtv, totMD);
    // selective scan + D-skip + silu(z) gate -> y16
    k_scan<<<(totSD + 255) / 256, 256, 0, stream>>>(
        dtv, dbc, xc, xz, A_log + (size_t)i * DIN * DS, Dp + (size_t)i * DIN, y16);
    // p = y @ out_w^T          (N=128, K=256), stored f16 for the next GEMM
    launch_gemm<_Float16>(y16, ow16 + (size_t)i * DM * DIN, p16,
                          MROWS, DM, DIN, DIN, DIN, DM, stream);
  }

  // head: (512, 96) = flat(512, 8320) @ head_w^T
  launch_gemm<float>(p16, hw16, headb, NSEQ, PRED, HNF, HNF, HNF, PRED, stream);

  const int totOut = BB * PRED * CC;
  k_final<<<(totOut + 255) / 256, 256, 0, stream>>>(headb, ms, out);
}
